// Attention_9844065042666
// MI455X (gfx1250) — matrix operand order repaired and checked by reading, not compiled, not run on hardware
//
#include <hip/hip_runtime.h>

#define DEVI __device__ __forceinline__

namespace {
constexpr int B = 4, S = 1024, D = 768, N = 12, H = 64;
constexpr int NH = N * H;            // 768
constexpr float QSCALE = 0.125f;     // 1/sqrt(64)
constexpr float NEG_BIG = -1.0e30f;
constexpr int PLD = 48;              // padded LDS row stride (elements) for P tiles
}

typedef __attribute__((ext_vector_type(16))) __bf16 v16bf;
typedef __attribute__((ext_vector_type(8)))  float  v8f;
typedef __attribute__((ext_vector_type(4)))  unsigned int u32x4;

union Frag { u32x4 q[2]; v16bf v; };

DEVI unsigned short f2bf(float f) {
  __bf16 h = (__bf16)f;
  return __builtin_bit_cast(unsigned short, h);
}

DEVI v8f zero8() {
  v8f z = {0.f, 0.f, 0.f, 0.f, 0.f, 0.f, 0.f, 0.f};
  return z;
}

DEVI v8f wmma_bf16(v16bf a, v16bf b, v8f c) {
  // D = A(16x32 bf16) * B(32x16 bf16) + C(16x16 f32)
  return __builtin_amdgcn_wmma_f32_16x16x32_bf16(false, a, false, b, (short)0, c,
                                                 false, false);
}

// A fragment (16x32, bf16) from row-major [M][K] storage.
// lane: row = lane%16; two contiguous 8-element chunks at k0+half*8 and k0+16+half*8.
DEVI v16bf load_a(const unsigned short* base, int ld, int m0, int k0, int lane) {
  const int r  = m0 + (lane & 15);
  const int kk = k0 + ((lane >> 4) << 3);
  const unsigned short* p = base + (size_t)r * ld + kk;
  Frag f;
  f.q[0] = *(const u32x4*)(p);
  f.q[1] = *(const u32x4*)(p + 16);
  return f.v;
}

// B fragment (32x16, bf16) from [Ncol][K] ("transposed") storage: row n is the
// output column; two contiguous 8-element chunks at k0+half*8 and k0+16+half*8.
DEVI v16bf load_b(const unsigned short* base, int ld, int k0, int n0, int lane) {
  const int n  = n0 + (lane & 15);
  const int kk = k0 + ((lane >> 4) << 3);
  const unsigned short* p = base + (size_t)n * ld + kk;
  Frag f;
  f.q[0] = *(const u32x4*)(p);
  f.q[1] = *(const u32x4*)(p + 16);
  return f.v;
}

// ---------------- prep kernels ----------------

__global__ void cvt_bf16_kernel(const float* __restrict__ in,
                                unsigned short* __restrict__ out, int n) {
  int i = blockIdx.x * blockDim.x + threadIdx.x;
  if (i < n) out[i] = f2bf(in[i]);
}

// out[b][c][r] = in[b][r][c]  (batched transpose, fp32 -> bf16)
__global__ void transpose_bf16_kernel(const float* __restrict__ in,
                                      unsigned short* __restrict__ out,
                                      int batch, int rows, int cols) {
  int i = blockIdx.x * blockDim.x + threadIdx.x;
  int total = batch * rows * cols;
  if (i >= total) return;
  int c  = i % cols;
  int r  = (i / cols) % rows;
  int bb = i / (cols * rows);
  out[((size_t)bb * cols + c) * rows + r] = f2bf(in[i]);
}

// ---------------- QKV projection ----------------
// One wave per (b, head, 32-row s tile): computes 32x64 tiles of Q, K, V.
// Two A fragments share every weight B fragment (24 WMMAs / 14 loads per K step).
// Q stored [b][n][s][h] pre-scaled; K stored [b][n][s][h]; V stored [b][n][h][s].
__global__ __launch_bounds__(128) void qkv_gemm_kernel(
    const unsigned short* __restrict__ xbf,  // [B][S][D]
    const unsigned short* __restrict__ wqt,  // [N][H][D]
    const unsigned short* __restrict__ wkt,  // [N][H][D]
    const unsigned short* __restrict__ wvt,  // [N][H][D]
    const float* __restrict__ bQ,            // [N][H]
    const float* __restrict__ bK,
    const float* __restrict__ bV,
    unsigned short* __restrict__ Qb,         // [B][N][S][H]
    unsigned short* __restrict__ Kb,         // [B][N][S][H]
    unsigned short* __restrict__ Vt)         // [B][N][H][S]
{
  const int lane = threadIdx.x & 31;
  const int wid  = blockIdx.x * (blockDim.x >> 5) + (threadIdx.x >> 5);
  const int stile = wid % (S / 32);
  const int tmp   = wid / (S / 32);
  const int head  = tmp % N;
  const int b     = tmp / N;
  const int m0 = stile * 32;

  const unsigned short* xrow = xbf + (size_t)b * S * D;
  const unsigned short* wq = wqt + (size_t)head * H * D;
  const unsigned short* wk = wkt + (size_t)head * H * D;
  const unsigned short* wv = wvt + (size_t)head * H * D;

  v8f aq[2][4], ak[2][4], av[2][4];
#pragma unroll
  for (int mi = 0; mi < 2; ++mi)
#pragma unroll
    for (int j = 0; j < 4; ++j) {
      aq[mi][j] = zero8(); ak[mi][j] = zero8(); av[mi][j] = zero8();
    }

  for (int k0 = 0; k0 < D; k0 += 32) {
    v16bf a0 = load_a(xrow, D, m0,      k0, lane);
    v16bf a1 = load_a(xrow, D, m0 + 16, k0, lane);
#pragma unroll
    for (int j = 0; j < 4; ++j) {
      const v16bf bq = load_b(wq, D, k0, j * 16, lane);
      aq[0][j] = wmma_bf16(a0, bq, aq[0][j]);
      aq[1][j] = wmma_bf16(a1, bq, aq[1][j]);
      const v16bf bk = load_b(wk, D, k0, j * 16, lane);
      ak[0][j] = wmma_bf16(a0, bk, ak[0][j]);
      ak[1][j] = wmma_bf16(a1, bk, ak[1][j]);
      const v16bf bv = load_b(wv, D, k0, j * 16, lane);
      av[0][j] = wmma_bf16(a0, bv, av[0][j]);
      av[1][j] = wmma_bf16(a1, bv, av[1][j]);
    }
  }

  const int half = lane >> 4, col = lane & 15;
  const size_t bn = (size_t)(b * N + head);
#pragma unroll
  for (int j = 0; j < 4; ++j) {
    const int h = j * 16 + col;
    const float biq = bQ[head * H + h];
    const float bik = bK[head * H + h];
    const float biv = bV[head * H + h];
#pragma unroll
    for (int mi = 0; mi < 2; ++mi) {
#pragma unroll
      for (int r = 0; r < 8; ++r) {
        const int m = m0 + mi * 16 + r + 8 * half;
        Qb[(bn * S + m) * H + h] = f2bf((aq[mi][j][r] + biq) * QSCALE);
        Kb[(bn * S + m) * H + h] = f2bf(ak[mi][j][r] + bik);
        Vt[(bn * H + h) * S + m] = f2bf(av[mi][j][r] + biv);
      }
    }
  }
}

// ---------------- flash attention ----------------
// One wave per (b, head, 16-row q tile). Online softmax over causal k tiles of 32.
__global__ __launch_bounds__(128) void attn_kernel(
    const unsigned short* __restrict__ Qb,  // [B][N][S][H] (pre-scaled)
    const unsigned short* __restrict__ Kb,  // [B][N][S][H]
    const unsigned short* __restrict__ Vt,  // [B][N][H][S]
    unsigned short* __restrict__ Zb)        // [B][S][N*H]
{
  __shared__ unsigned short pbuf[4][16 * PLD];

  const int lane  = threadIdx.x & 31;
  const int wslot = threadIdx.x >> 5;
  const int wid   = blockIdx.x * 4 + wslot;
  const int qt   = wid % (S / 16);
  const int tmp  = wid / (S / 16);
  const int head = tmp % N;
  const int b    = tmp / N;
  const int m0 = qt * 16;

  const size_t bn = (size_t)(b * N + head);
  const unsigned short* qbase = Qb + bn * S * H;
  const unsigned short* kbase = Kb + bn * S * H;
  const unsigned short* vbase = Vt + bn * H * S;

  const v16bf qa0 = load_a(qbase, H, m0, 0, lane);
  const v16bf qa1 = load_a(qbase, H, m0, 32, lane);

  v8f o[4];
#pragma unroll
  for (int j = 0; j < 4; ++j) o[j] = zero8();
  float rmax[8], rsum[8];
#pragma unroll
  for (int r = 0; r < 8; ++r) { rmax[r] = NEG_BIG; rsum[r] = 0.f; }

  const int half = lane >> 4, col = lane & 15;
  unsigned short* pl = pbuf[wslot];
  const int nsteps = m0 / 32 + 1;

  for (int st = 0; st < nsteps; ++st) {
    const int sk0 = st * 32;

    // scores: two 16x16 C tiles covering keys [sk0, sk0+32)
    v8f s0 = zero8(), s1 = zero8();
    s0 = wmma_bf16(qa0, load_b(kbase, H, 0,  sk0,      lane), s0);
    s0 = wmma_bf16(qa1, load_b(kbase, H, 32, sk0,      lane), s0);
    s1 = wmma_bf16(qa0, load_b(kbase, H, 0,  sk0 + 16, lane), s1);
    s1 = wmma_bf16(qa1, load_b(kbase, H, 32, sk0 + 16, lane), s1);

    if (sk0 + 31 > m0) {  // diagonal tile: apply causal mask
#pragma unroll
      for (int r = 0; r < 8; ++r) {
        const int q = m0 + r + 8 * half;
        if (sk0 + col > q)      s0[r] = NEG_BIG;
        if (sk0 + 16 + col > q) s1[r] = NEG_BIG;
      }
    }

    // online softmax, P tile to LDS (row-major, bf16)
#pragma unroll
    for (int r = 0; r < 8; ++r) {
      float mloc = fmaxf(s0[r], s1[r]);
#pragma unroll
      for (int off = 8; off > 0; off >>= 1) mloc = fmaxf(mloc, __shfl_xor(mloc, off, 32));
      const float mnew = fmaxf(rmax[r], mloc);
      const float corr = __expf(rmax[r] - mnew);
      const float p0 = __expf(s0[r] - mnew);
      const float p1 = __expf(s1[r] - mnew);
      float rs = p0 + p1;
#pragma unroll
      for (int off = 8; off > 0; off >>= 1) rs += __shfl_xor(rs, off, 32);
      rsum[r] = rsum[r] * corr + rs;
      rmax[r] = mnew;
#pragma unroll
      for (int j = 0; j < 4; ++j) o[j][r] *= corr;
      const int row = r + 8 * half;
      pl[row * PLD + col]      = f2bf(p0);
      pl[row * PLD + 16 + col] = f2bf(p1);
    }

    // cross-lane RAW through LDS: wait for the DS stores, then re-fragment P
    asm volatile("s_wait_dscnt 0" ::: "memory");
    const v16bf pa = load_a(pl, PLD, 0, 0, lane);

#pragma unroll
    for (int j = 0; j < 4; ++j)
      o[j] = wmma_bf16(pa, load_b(vbase, S, sk0, j * 16, lane), o[j]);
  }

  // normalize and emit z[b][q][head*H + h] (bf16, contiguous nh for final GEMM)
#pragma unroll
  for (int j = 0; j < 4; ++j) {
#pragma unroll
    for (int r = 0; r < 8; ++r) {
      const int q = m0 + r + 8 * half;
      Zb[((size_t)b * S + q) * NH + head * H + j * 16 + col] = f2bf(o[j][r] / rsum[r]);
    }
  }
}

// ---------------- output projection ----------------
// One wave per (32-row tile of B*S, 64-col tile of D); 2 A-frags share B-frags.
__global__ __launch_bounds__(128) void out_gemm_kernel(
    const unsigned short* __restrict__ Zb,   // [B*S][NH]
    const unsigned short* __restrict__ wot,  // [D][NH]
    const float* __restrict__ bO,            // [D]
    float* __restrict__ out)                 // [B*S][D]
{
  const int lane = threadIdx.x & 31;
  const int wid  = blockIdx.x * (blockDim.x >> 5) + (threadIdx.x >> 5);
  const int dt = wid % (D / 64);
  const int mt = wid / (D / 64);
  const int m0 = mt * 32, d0 = dt * 64;

  v8f acc[2][4];
#pragma unroll
  for (int mi = 0; mi < 2; ++mi)
#pragma unroll
    for (int j = 0; j < 4; ++j) acc[mi][j] = zero8();

  for (int k0 = 0; k0 < NH; k0 += 32) {
    v16bf a0 = load_a(Zb, NH, m0,      k0, lane);
    v16bf a1 = load_a(Zb, NH, m0 + 16, k0, lane);
#pragma unroll
    for (int j = 0; j < 4; ++j) {
      const v16bf bfr = load_b(wot, NH, k0, d0 + j * 16, lane);
      acc[0][j] = wmma_bf16(a0, bfr, acc[0][j]);
      acc[1][j] = wmma_bf16(a1, bfr, acc[1][j]);
    }
  }

  const int half = lane >> 4, col = lane & 15;
#pragma unroll
  for (int j = 0; j < 4; ++j) {
    const float bias = bO[d0 + j * 16 + col];
#pragma unroll
    for (int mi = 0; mi < 2; ++mi) {
#pragma unroll
      for (int r = 0; r < 8; ++r) {
        const int m = m0 + mi * 16 + r + 8 * half;
        out[(size_t)m * D + d0 + j * 16 + col] = acc[mi][j][r] + bias;
      }
    }
  }
}

// ---------------- host ----------------

extern "C" void kernel_launch(void* const* d_in, const int* in_sizes, int n_in,
                              void* d_out, int out_size, void* d_ws, size_t ws_size,
                              hipStream_t stream) {
  (void)in_sizes; (void)n_in; (void)out_size; (void)ws_size;
  const float* x  = (const float*)d_in[0];
  const float* WQ = (const float*)d_in[1];
  const float* WK = (const float*)d_in[2];
  const float* WV = (const float*)d_in[3];
  const float* WO = (const float*)d_in[4];
  const float* bQ = (const float*)d_in[5];
  const float* bK = (const float*)d_in[6];
  const float* bV = (const float*)d_in[7];
  const float* bO = (const float*)d_in[8];
  float* out = (float*)d_out;

  char* ws = (char*)d_ws;
  size_t off = 0;
  auto take = [&](size_t elems) {
    unsigned short* p = (unsigned short*)(ws + off);
    off += (elems * 2 + 255) & ~(size_t)255;
    return p;
  };
  unsigned short* xbf = take((size_t)B * S * D);
  unsigned short* wqt = take((size_t)N * H * D);
  unsigned short* wkt = take((size_t)N * H * D);
  unsigned short* wvt = take((size_t)N * H * D);
  unsigned short* wot = take((size_t)D * NH);
  unsigned short* Qb  = take((size_t)B * N * S * H);
  unsigned short* Kb  = take((size_t)B * N * S * H);
  unsigned short* Vt  = take((size_t)B * N * S * H);
  unsigned short* Zb  = take((size_t)B * S * NH);

  const int nx = B * S * D;
  cvt_bf16_kernel<<<(nx + 255) / 256, 256, 0, stream>>>(x, xbf, nx);

  const int nw = N * D * H;
  transpose_bf16_kernel<<<(nw + 255) / 256, 256, 0, stream>>>(WQ, wqt, N, D, H);
  transpose_bf16_kernel<<<(nw + 255) / 256, 256, 0, stream>>>(WK, wkt, N, D, H);
  transpose_bf16_kernel<<<(nw + 255) / 256, 256, 0, stream>>>(WV, wvt, N, D, H);
  transpose_bf16_kernel<<<(nw + 255) / 256, 256, 0, stream>>>(WO, wot, 1, NH, D);

  const int waves_qkv = B * N * (S / 32);             // 1536
  qkv_gemm_kernel<<<waves_qkv / 4, 128, 0, stream>>>(xbf, wqt, wkt, wvt,
                                                     bQ, bK, bV, Qb, Kb, Vt);

  const int waves_attn = B * N * (S / 16);            // 3072
  attn_kernel<<<waves_attn / 4, 128, 0, stream>>>(Qb, Kb, Vt, Zb);

  const int waves_out = B * (S / 32) * (D / 64);      // 1536
  out_gemm_kernel<<<waves_out / 4, 128, 0, stream>>>(Zb, wot, bO, out);
}
